// AttentionForONNX_12867722019658
// MI455X (gfx1250) — matrix operand order repaired and checked by reading, not compiled, not run on hardware
//
#include <hip/hip_runtime.h>
#include <hip/hip_bf16.h>

#define B_  16
#define T_  512
#define E_  512
#define H_  8
#define HD_ 64
#define S2_ 1023
#define MQ_ (B_ * T_)    // 8192 rows
#define MP_ (B_ * S2_)   // 16368 rows (multiple of 16)

typedef __attribute__((ext_vector_type(16))) __bf16 v16bf;
typedef __attribute__((ext_vector_type(8)))  float  v8f;

union FragAB {
  v16bf v;
  unsigned short us[16];
  uint4 q[2];
};

// hardware convert (v_cvt bf16) instead of manual bit twiddling
__device__ __forceinline__ unsigned short f2bf(float f) {
  __bf16 h = (__bf16)f;
  return __builtin_bit_cast(unsigned short, h);
}

// ------------------------------------------- vectorized fp32 -> bf16 (8/thread)
__global__ __launch_bounds__(256)
void cvt_bf16x8_kernel(const float* __restrict__ src,
                       unsigned short* __restrict__ dst, int n8) {
  int i = blockIdx.x * blockDim.x + threadIdx.x;
  if (i >= n8) return;
  const float4 a = *reinterpret_cast<const float4*>(src + (size_t)i * 8);
  const float4 b = *reinterpret_cast<const float4*>(src + (size_t)i * 8 + 4);
  union { uint4 q; unsigned short us[8]; } o;
  o.us[0] = f2bf(a.x); o.us[1] = f2bf(a.y); o.us[2] = f2bf(a.z); o.us[3] = f2bf(a.w);
  o.us[4] = f2bf(b.x); o.us[5] = f2bf(b.y); o.us[6] = f2bf(b.z); o.us[7] = f2bf(b.w);
  *reinterpret_cast<uint4*>(dst + (size_t)i * 8) = o.q;
}

// -------------------------------------------------- projection GEMM via WMMA
// C[M,512] = A[M,512] (bf16) @ W^T, W bf16 row-major [N=512,K=512]
// (row-major W is directly the WMMA B-fragment source, since B[k,n]=W[n,k]).
// mode 0: out0=q+b+pbu, out1=q+b+pbv (bf16 row-major)
// mode 2: v -> out0 transposed [B,H,HD,T] bf16
// else  : out0 bf16 row-major (+bias if non-null)
__global__ __launch_bounds__(256)
void proj_kernel(const unsigned short* __restrict__ Ab, int M,
                 const unsigned short* __restrict__ Wb,
                 const float* __restrict__ bias,
                 const float* __restrict__ pbu,
                 const float* __restrict__ pbv,
                 unsigned short* __restrict__ out0,
                 unsigned short* __restrict__ out1,
                 int mode) {
  const int lane = threadIdx.x & 31;
  const int wid  = blockIdx.x * (blockDim.x >> 5) + (threadIdx.x >> 5);
  const int ntiles = (M >> 4) * (E_ >> 4);
  if (wid >= ntiles) return;                 // wave-uniform: EXEC stays all-1s
  const int tm = wid >> 5;                   // 32 column tiles (E/16)
  const int tn = wid & 31;
  const int nh   = lane & 15;
  const int half = lane >> 4;

  v8f acc = {0.f, 0.f, 0.f, 0.f, 0.f, 0.f, 0.f, 0.f};

  const unsigned short* arow = Ab + (size_t)(tm * 16 + nh) * E_;
  const unsigned short* wrow = Wb + (size_t)(tn * 16 + nh) * E_;

#pragma unroll 4
  for (int kk = 0; kk < E_; kk += 32) {
    FragAB af, bf;
    // A frag: lane row M=l%16, 16 contiguous K at (l/16)*16 (two b128 loads)
    const unsigned short* ap = arow + kk + half * 16;
    af.q[0] = *reinterpret_cast<const uint4*>(ap);
    af.q[1] = *reinterpret_cast<const uint4*>(ap + 8);
    // B frag: lane col N=l%16, 16 contiguous K at (l/16)*16
    const unsigned short* bp = wrow + kk + half * 16;
    bf.q[0] = *reinterpret_cast<const uint4*>(bp);
    bf.q[1] = *reinterpret_cast<const uint4*>(bp + 8);
    if (kk + 32 < E_) {
      __builtin_prefetch(ap + 32, 0, 0);
      __builtin_prefetch(bp + 32, 0, 0);
    }
    acc = __builtin_amdgcn_wmma_f32_16x16x32_bf16(false, af.v, false, bf.v,
                                                  (short)0, acc, false, false);
  }

  const int e  = tn * 16 + nh;
  const float bb = bias ? bias[e] : 0.f;
  if (mode == 0) {
    const float u = pbu[e], vv = pbv[e];
#pragma unroll
    for (int j = 0; j < 8; ++j) {
      const int m = tm * 16 + j + 8 * half;
      const float q = acc[j] + bb;
      out0[(size_t)m * E_ + e] = f2bf(q + u);
      out1[(size_t)m * E_ + e] = f2bf(q + vv);
    }
  } else if (mode == 2) {
    const int hh = e >> 6, dd = e & 63;
#pragma unroll
    for (int j = 0; j < 8; ++j) {
      const int m = tm * 16 + j + 8 * half;
      const int bi = m >> 9, ti = m & (T_ - 1);
      out0[(((size_t)(bi * H_ + hh)) * HD_ + dd) * T_ + ti] = f2bf(acc[j] + bb);
    }
  } else {
#pragma unroll
    for (int j = 0; j < 8; ++j) {
      const int m = tm * 16 + j + 8 * half;
      out0[(size_t)m * E_ + e] = f2bf(acc[j] + bb);
    }
  }
}

// --------------------------------------- fused rel-pos attention (flash-style)
#define WPB 4   // independent waves per block, one 16-row query tile each

__global__ __launch_bounds__(32 * WPB)
void relattn_kernel(const unsigned short* __restrict__ qu,  // [MQ,E] bf16
                    const unsigned short* __restrict__ qv,  // [MQ,E] bf16
                    const unsigned short* __restrict__ kmat,// [MQ,E] bf16
                    const unsigned short* __restrict__ pmat,// [MP,E] bf16
                    const unsigned short* __restrict__ vT,  // [B,H,HD,T] bf16
                    float* __restrict__ out)                // [B,T,E] fp32
{
  __shared__ float          sbd[WPB][16 * 48];   // raw-bd band for rel-shift
  __shared__ unsigned short spr[WPB][16 * 32];   // probs transpose staging

  const int lane = threadIdx.x & 31;
  const int w    = threadIdx.x >> 5;
  const int tt   = blockIdx.x * WPB + w;
  const int h    = blockIdx.y;
  const int b    = blockIdx.z;
  const int t0   = tt * 16;
  const int nh   = lane & 15;
  const int half = lane >> 4;

  // Preload q_u / q_v A-fragments for K = 0..63 (reused for every s-step).
  FragAB au[2], av[2];
  {
    const unsigned short* qur = qu + ((size_t)(b * T_ + t0 + nh)) * E_ + h * HD_;
    const unsigned short* qvr = qv + ((size_t)(b * T_ + t0 + nh)) * E_ + h * HD_;
#pragma unroll
    for (int ks = 0; ks < 2; ++ks) {
      const int kb = ks * 32 + half * 16;
      au[ks].q[0] = *reinterpret_cast<const uint4*>(qur + kb);
      au[ks].q[1] = *reinterpret_cast<const uint4*>(qur + kb + 8);
      av[ks].q[0] = *reinterpret_cast<const uint4*>(qvr + kb);
      av[ks].q[1] = *reinterpret_cast<const uint4*>(qvr + kb + 8);
    }
  }

  float mrow[8], lrow[8];
  v8f oacc[4];
  const v8f z = {0.f, 0.f, 0.f, 0.f, 0.f, 0.f, 0.f, 0.f};
#pragma unroll
  for (int j = 0; j < 8; ++j) { mrow[j] = -__builtin_inff(); lrow[j] = 0.f; }
#pragma unroll
  for (int d = 0; d < 4; ++d) oacc[d] = z;

  for (int s0 = 0; s0 < T_; s0 += 32) {
    // ---- content scores ac: 16x32 (two 16x16 tiles) ----
    v8f sc0 = z, sc1 = z;
#pragma unroll
    for (int ks = 0; ks < 2; ++ks) {
      const int kb = ks * 32 + half * 16;
      FragAB bk0, bk1;
      const unsigned short* k0 = kmat + ((size_t)(b * T_ + s0 + nh)) * E_ + h * HD_ + kb;
      const unsigned short* k1 = kmat + ((size_t)(b * T_ + s0 + 16 + nh)) * E_ + h * HD_ + kb;
      bk0.q[0] = *reinterpret_cast<const uint4*>(k0);
      bk0.q[1] = *reinterpret_cast<const uint4*>(k0 + 8);
      bk1.q[0] = *reinterpret_cast<const uint4*>(k1);
      bk1.q[1] = *reinterpret_cast<const uint4*>(k1 + 8);
      sc0 = __builtin_amdgcn_wmma_f32_16x16x32_bf16(false, au[ks].v, false, bk0.v,
                                                    (short)0, sc0, false, false);
      sc1 = __builtin_amdgcn_wmma_f32_16x16x32_bf16(false, au[ks].v, false, bk1.v,
                                                    (short)0, sc1, false, false);
    }
    if (s0 + 32 < T_) {   // hint next s-step K rows toward L2/L0
      __builtin_prefetch(kmat + ((size_t)(b * T_ + s0 + 32 + nh)) * E_ + h * HD_, 0, 0);
      __builtin_prefetch(kmat + ((size_t)(b * T_ + s0 + 48 + nh)) * E_ + h * HD_, 0, 0);
    }

    // ---- positional scores: raw band bd[t, p0..p0+47], p0 = s0+T-1-t0-15 ----
    const int p0 = s0 + (T_ - 1) - t0 - 15;   // >= 0 for all tiles
    v8f bd[3] = {z, z, z};
#pragma unroll
    for (int c = 0; c < 3; ++c) {
      int prow = p0 + c * 16 + nh;
      prow = prow < (S2_ - 1) ? prow : (S2_ - 1);   // clamp one-past corner
      const unsigned short* pr = pmat + ((size_t)(b * S2_ + prow)) * E_ + h * HD_;
#pragma unroll
      for (int ks = 0; ks < 2; ++ks) {
        const int kb = ks * 32 + half * 16;
        FragAB bp;
        bp.q[0] = *reinterpret_cast<const uint4*>(pr + kb);
        bp.q[1] = *reinterpret_cast<const uint4*>(pr + kb + 8);
        bd[c] = __builtin_amdgcn_wmma_f32_16x16x32_bf16(false, av[ks].v, false, bp.v,
                                                        (short)0, bd[c], false, false);
      }
    }

    // rel-shift through LDS: shifted[t,s] = band[t-t0, 15 + (s-s0) - (t-t0)]
    float* bdl = sbd[w];
#pragma unroll
    for (int c = 0; c < 3; ++c)
#pragma unroll
      for (int j = 0; j < 8; ++j)
        bdl[(j + 8 * half) * 48 + c * 16 + nh] = bd[c][j];
    asm volatile("s_wait_dscnt 0" ::: "memory");
#pragma unroll
    for (int j = 0; j < 8; ++j) {
      const int m = j + 8 * half;
      sc0[j] = (sc0[j] + bdl[m * 48 + 15 + nh - m]) * 0.125f;        // 1/sqrt(64)
      sc1[j] = (sc1[j] + bdl[m * 48 + 31 + nh - m]) * 0.125f;
    }

    // ---- online softmax (16-lane butterfly = one score row) ----
    float pr0[8], pr1[8], fac[8];
#pragma unroll
    for (int j = 0; j < 8; ++j) {
      float mx = fmaxf(sc0[j], sc1[j]);
#pragma unroll
      for (int off = 1; off < 16; off <<= 1) mx = fmaxf(mx, __shfl_xor(mx, off, 32));
      const float mnew = fmaxf(mrow[j], mx);
      const float e0 = __expf(sc0[j] - mnew);
      const float e1 = __expf(sc1[j] - mnew);
      float s = e0 + e1;
#pragma unroll
      for (int off = 1; off < 16; off <<= 1) s += __shfl_xor(s, off, 32);
      const float f = __expf(mrow[j] - mnew);
      mrow[j] = mnew;
      lrow[j] = lrow[j] * f + s;
      fac[j] = f; pr0[j] = e0; pr1[j] = e1;
    }
#pragma unroll
    for (int d = 0; d < 4; ++d)
#pragma unroll
      for (int j = 0; j < 8; ++j) oacc[d][j] *= fac[j];

    // ---- probs (C-layout) -> LDS -> bf16 A-fragment ----
    unsigned short* prl = spr[w];
#pragma unroll
    for (int j = 0; j < 8; ++j) {
      const int m = j + 8 * half;
      prl[m * 32 + nh]      = f2bf(pr0[j]);
      prl[m * 32 + 16 + nh] = f2bf(pr1[j]);
    }
    asm volatile("s_wait_dscnt 0" ::: "memory");
    FragAB pf;
    {
      const int kb = half * 16;
#pragma unroll
      for (int i = 0; i < 8; ++i) {
        pf.us[i]     = prl[nh * 32 + kb + i];
        pf.us[8 + i] = prl[nh * 32 + kb + 8 + i];
      }
    }

    // ---- PV: out[16,64] += probs[16,32] @ v[32,64] (vT gives B-fragments) ----
#pragma unroll
    for (int d = 0; d < 4; ++d) {
      FragAB bv_;
      const unsigned short* vp =
          vT + (((size_t)(b * H_ + h)) * HD_ + d * 16 + nh) * T_ + s0 + half * 16;
      bv_.q[0] = *reinterpret_cast<const uint4*>(vp);
      bv_.q[1] = *reinterpret_cast<const uint4*>(vp + 8);
      oacc[d] = __builtin_amdgcn_wmma_f32_16x16x32_bf16(false, pf.v, false, bv_.v,
                                                        (short)0, oacc[d], false, false);
    }
  }

  // ---- epilogue: divide by softmax denominator, store fp32 ----
#pragma unroll
  for (int d = 0; d < 4; ++d)
#pragma unroll
    for (int j = 0; j < 8; ++j) {
      const int m = t0 + j + 8 * half;
      out[((size_t)(b * T_ + m)) * E_ + h * HD_ + d * 16 + nh] = oacc[d][j] / lrow[j];
    }
}

// ------------------------------------------------------------------ launcher
extern "C" void kernel_launch(void* const* d_in, const int* in_sizes, int n_in,
                              void* d_out, int out_size, void* d_ws, size_t ws_size,
                              hipStream_t stream) {
  const float* x   = (const float*)d_in[0];
  const float* pos = (const float*)d_in[1];
  const float* Wq  = (const float*)d_in[2];
  const float* bq  = (const float*)d_in[3];
  const float* Wk  = (const float*)d_in[4];
  const float* bk  = (const float*)d_in[5];
  const float* Wv  = (const float*)d_in[6];
  const float* bv  = (const float*)d_in[7];
  const float* Wp  = (const float*)d_in[8];
  const float* pbu = (const float*)d_in[9];
  const float* pbv = (const float*)d_in[10];
  float* out = (float*)d_out;

  char* ws = (char*)d_ws;
  unsigned short* Wq_b  = (unsigned short*)ws; ws += (size_t)E_ * E_ * 2;
  unsigned short* Wk_b  = (unsigned short*)ws; ws += (size_t)E_ * E_ * 2;
  unsigned short* Wv_b  = (unsigned short*)ws; ws += (size_t)E_ * E_ * 2;
  unsigned short* Wp_b  = (unsigned short*)ws; ws += (size_t)E_ * E_ * 2;
  unsigned short* x_b   = (unsigned short*)ws; ws += (size_t)MQ_ * E_ * 2;
  unsigned short* pos_b = (unsigned short*)ws; ws += (size_t)MP_ * E_ * 2;
  unsigned short* qu_b  = (unsigned short*)ws; ws += (size_t)MQ_ * E_ * 2;
  unsigned short* qv_b  = (unsigned short*)ws; ws += (size_t)MQ_ * E_ * 2;
  unsigned short* k_b   = (unsigned short*)ws; ws += (size_t)MQ_ * E_ * 2;
  unsigned short* p_b   = (unsigned short*)ws; ws += (size_t)MP_ * E_ * 2;
  unsigned short* vT_b  = (unsigned short*)ws; ws += (size_t)MQ_ * E_ * 2;

  // one-time fp32 -> bf16 conversions (8 elems/thread, b128 in / b128 out)
  const int wn8 = (E_ * E_) / 8;            // 32768
  const int xn8 = (MQ_ * E_) / 8;           // 524288
  const int pn8 = (MP_ * E_) / 8;           // 1047552
  cvt_bf16x8_kernel<<<(wn8 + 255) / 256, 256, 0, stream>>>(Wq, Wq_b, wn8);
  cvt_bf16x8_kernel<<<(wn8 + 255) / 256, 256, 0, stream>>>(Wk, Wk_b, wn8);
  cvt_bf16x8_kernel<<<(wn8 + 255) / 256, 256, 0, stream>>>(Wv, Wv_b, wn8);
  cvt_bf16x8_kernel<<<(wn8 + 255) / 256, 256, 0, stream>>>(Wp, Wp_b, wn8);
  cvt_bf16x8_kernel<<<(xn8 + 255) / 256, 256, 0, stream>>>(x,  x_b,  xn8);
  cvt_bf16x8_kernel<<<(pn8 + 255) / 256, 256, 0, stream>>>(pos, pos_b, pn8);

  const int tilesQ = (MQ_ / 16) * (E_ / 16);   // 16384 -> 2048 blocks of 8 waves
  const int tilesP = (MP_ / 16) * (E_ / 16);   // 32736 -> 4092 blocks
  proj_kernel<<<tilesQ / 8, 256, 0, stream>>>(x_b, MQ_, Wq_b, bq, pbu, pbv, qu_b, qv_b, 0);
  proj_kernel<<<tilesQ / 8, 256, 0, stream>>>(x_b, MQ_, Wk_b, bk, nullptr, nullptr, k_b, nullptr, 1);
  proj_kernel<<<tilesQ / 8, 256, 0, stream>>>(x_b, MQ_, Wv_b, bv, nullptr, nullptr, vT_b, nullptr, 2);
  proj_kernel<<<tilesP / 8, 256, 0, stream>>>(pos_b, MP_, Wp_b, nullptr, nullptr, nullptr, p_b, nullptr, 3);

  dim3 grid(T_ / 16 / WPB, H_, B_);            // (8, 8, 16)
  relattn_kernel<<<grid, 32 * WPB, 0, stream>>>(qu_b, qv_b, k_b, p_b, vT_b, out);
}
